// Discriminator_24008867185216
// MI455X (gfx1250) — compile-verified, matrix operand order checked by reading, not run on hardware
//
#include <hip/hip_runtime.h>
#include <hip/hip_bf16.h>

#define N_NODES 50000
#define N_EDGES 800000
#define N_GRAPH 64
#define D_IN    128
#define HID     512
#define SN_ITERS 5

typedef __attribute__((ext_vector_type(16))) __bf16 v16bf;
typedef __attribute__((ext_vector_type(4)))  __bf16 v4bf;
typedef __attribute__((ext_vector_type(8)))  float  v8f;

#define AS_GLOBAL __attribute__((address_space(1)))
#define AS_LDS    __attribute__((address_space(3)))

// Exact pointee type expected by the async-to-LDS builtin (per hipcc diagnostic):
// 'int __attribute__((vector_size(16)))'
typedef int v4i_vec __attribute__((vector_size(16)));
typedef AS_GLOBAL v4i_vec* async_gptr_t;
typedef AS_LDS    v4i_vec* async_lptr_t;

#if defined(__has_builtin)
#if __has_builtin(__builtin_amdgcn_global_load_async_to_lds_b128) && \
    __has_builtin(__builtin_amdgcn_s_wait_asynccnt)
#define USE_ASYNC_LDS 1
#endif
#if __has_builtin(__builtin_amdgcn_sched_group_barrier)
#define HAVE_SCHED_HINT 1
#endif
#endif

#if defined(HAVE_SCHED_HINT)
#define WMMA_SCHED_HINT()                                   \
    do {                                                    \
        __builtin_amdgcn_sched_group_barrier(0x020, 2, 0);  \
        __builtin_amdgcn_sched_group_barrier(0x100, 8, 0);  \
        __builtin_amdgcn_sched_group_barrier(0x008, 4, 0);  \
    } while (0)
#else
#define WMMA_SCHED_HINT() do { } while (0)
#endif

// ---------------------------------------------------------------------------
// Spectral normalization (5 power iterations) + scale + fp32->bf16 weight cast.
// One 512-thread block per weight matrix. sigma = ||W v_last|| (matches ref).
// ---------------------------------------------------------------------------
__global__ __launch_bounds__(512)
void sn_kernel(const float* __restrict__ W, int rows, int cols,
               __bf16* __restrict__ Wb) {
    __shared__ float u[HID];
    __shared__ float v[HID];
    __shared__ float red[512];
    const int t = threadIdx.x;
    if (t < rows) u[t] = rsqrtf((float)rows);
    __syncthreads();

    float tn = 1.0f;
    for (int it = 0; it < SN_ITERS; ++it) {
        // v = W^T u  (column dot)
        float acc = 0.0f;
        if (t < cols) {
            for (int i = 0; i < rows; ++i) acc += W[(size_t)i * cols + t] * u[i];
        }
        red[t] = acc * acc;
        __syncthreads();
        for (int s = 256; s > 0; s >>= 1) { if (t < s) red[t] += red[t + s]; __syncthreads(); }
        float inv = 1.0f / (sqrtf(red[0]) + 1e-12f);
        __syncthreads();
        if (t < cols) v[t] = acc * inv;
        __syncthreads();

        // u = W v  (row dot)
        float acc2 = 0.0f;
        if (t < rows) {
            const float* rw = W + (size_t)t * cols;
            for (int j = 0; j < cols; ++j) acc2 += rw[j] * v[j];
        }
        red[t] = acc2 * acc2;
        __syncthreads();
        for (int s = 256; s > 0; s >>= 1) { if (t < s) red[t] += red[t + s]; __syncthreads(); }
        tn = sqrtf(red[0]);                      // ||W v||  == sigma on last iter
        float inv2 = 1.0f / (tn + 1e-12f);
        __syncthreads();
        if (t < rows) u[t] = acc2 * inv2;
        __syncthreads();
    }

    const float invsig = 1.0f / tn;
    const int n = rows * cols;
    for (int i = t; i < n; i += 512) Wb[i] = (__bf16)(W[i] * invsig);
}

// ---------------------------------------------------------------------------
// agg = h  (fp32 copy, float4)
// ---------------------------------------------------------------------------
__global__ void copy_f4(const float4* __restrict__ s, float4* __restrict__ d, int n4) {
    int i = blockIdx.x * blockDim.x + threadIdx.x;
    if (i < n4) d[i] = s[i];
}

// ---------------------------------------------------------------------------
// agg[dst[e]] += h[src[e]]  : one thread handles one (edge, 4-float chunk).
// fp32 atomics -> global_atomic_add_f32 (no return).
// ---------------------------------------------------------------------------
__global__ void scatter_add(const float* __restrict__ h,
                            const int* __restrict__ srcI,
                            const int* __restrict__ dstI,
                            float* __restrict__ agg,
                            int K, int log2c) {
    long long idx = (long long)blockIdx.x * blockDim.x + threadIdx.x;
    long long total = (long long)N_EDGES << log2c;
    if (idx >= total) return;
    int e = (int)(idx >> log2c);
    int c = ((int)idx & ((1 << log2c) - 1)) << 2;
    const float4 val = *(const float4*)(h + (size_t)srcI[e] * K + c);
    float* p = agg + (size_t)dstI[e] * K + c;
    unsafeAtomicAdd(p + 0, val.x);
    unsafeAtomicAdd(p + 1, val.y);
    unsafeAtomicAdd(p + 2, val.z);
    unsafeAtomicAdd(p + 3, val.w);
}

// ---------------------------------------------------------------------------
// fp32 -> bf16 cast (4 wide)
// ---------------------------------------------------------------------------
__global__ void cast_f32_bf16(const float4* __restrict__ s, v4bf* __restrict__ d, int n4) {
    int i = blockIdx.x * blockDim.x + threadIdx.x;
    if (i < n4) {
        float4 f = s[i];
        v4bf o;
        o.x = (__bf16)f.x; o.y = (__bf16)f.y; o.z = (__bf16)f.z; o.w = (__bf16)f.w;
        d[i] = o;
    }
}

// ---------------------------------------------------------------------------
// WMMA GEMM: out[N,512] = act(A[N,K] @ W^T + bias)
//   A   : bf16 row-major [N,K]
//   Wb  : bf16 row-major [512,K]   (B = W^T, loaded as contiguous W rows)
// Block: 256 threads = 8 waves; block tile = 128 rows x 64 cols.
// Each wave: 16 rows x 64 cols = 4 x v_wmma_f32_16x16x32_bf16 per K-step.
// The 64-row weight panel (64*K bf16) is staged in LDS via async-to-LDS loads
// (ASYNCcnt) and shared by all 8 waves.
// ---------------------------------------------------------------------------
template <bool RELU, bool OUT_BF16>
__global__ __launch_bounds__(256)
void gemm_wmma(const __bf16* __restrict__ A, const __bf16* __restrict__ Wb,
               const float* __restrict__ bias, void* __restrict__ out,
               int nrows, int K) {
    extern __shared__ char smem[];          // 64 * K * 2 bytes, W-panel row-major
    const int tid  = threadIdx.x;
    const int lane = tid & 31;
    const int wave = tid >> 5;
    const int cBase = blockIdx.y * 64;

    // Stage the contiguous 64-row weight panel into LDS.
    {
        uint4* srcp = const_cast<uint4*>((const uint4*)(Wb + (size_t)cBase * K));
        uint4* dstp = (uint4*)smem;
        const int chunks = (64 * K) / 8;    // 16B chunks
#if defined(USE_ASYNC_LDS)
        for (int i = tid; i < chunks; i += 256) {
            __builtin_amdgcn_global_load_async_to_lds_b128(
                (async_gptr_t)(srcp + i), (async_lptr_t)(dstp + i), 0, 0);
        }
        __builtin_amdgcn_s_wait_asynccnt(0);
#else
        for (int i = tid; i < chunks; i += 256) dstp[i] = srcp[i];
#endif
    }
    __syncthreads();

    const int rowBase = blockIdx.x * 128 + wave * 16;
    const int m     = lane & 15;
    const int khalf = (lane >> 4) * 16;
    const int ar    = min(rowBase + m, nrows - 1);   // clamp: EXEC stays all-ones

    v8f acc0 = {0,0,0,0,0,0,0,0};
    v8f acc1 = {0,0,0,0,0,0,0,0};
    v8f acc2 = {0,0,0,0,0,0,0,0};
    v8f acc3 = {0,0,0,0,0,0,0,0};

    const size_t aRow = (size_t)ar * K;
    const size_t bRow = (size_t)m * K * 2;           // bytes
    const size_t bCt  = (size_t)16 * K * 2;          // bytes per 16-col tile

    for (int k = 0; k < K; k += 32) {
        v16bf a = *(const v16bf*)(A + aRow + k + khalf);
        const char* bp = smem + bRow + (size_t)(k + khalf) * 2;
        v16bf b0 = *(const v16bf*)(bp);
        v16bf b1 = *(const v16bf*)(bp + bCt);
        v16bf b2 = *(const v16bf*)(bp + 2 * bCt);
        v16bf b3 = *(const v16bf*)(bp + 3 * bCt);
        acc0 = __builtin_amdgcn_wmma_f32_16x16x32_bf16(false, a, false, b0, (short)0, acc0, false, false);
        acc1 = __builtin_amdgcn_wmma_f32_16x16x32_bf16(false, a, false, b1, (short)0, acc1, false, false);
        acc2 = __builtin_amdgcn_wmma_f32_16x16x32_bf16(false, a, false, b2, (short)0, acc2, false, false);
        acc3 = __builtin_amdgcn_wmma_f32_16x16x32_bf16(false, a, false, b3, (short)0, acc3, false, false);
        WMMA_SCHED_HINT();   // issue A vmem + 8 ds reads first, then 4 WMMAs
    }

    // D layout: lane l holds N = l&15; VGPR r holds M = r + 8*(l>>4).
    const int rbase = rowBase + 8 * (lane >> 4);
    v8f accs[4] = {acc0, acc1, acc2, acc3};
#pragma unroll
    for (int ct = 0; ct < 4; ++ct) {
        const int col = cBase + ct * 16 + m;
        const float bv = bias[col];
        v8f av = accs[ct];
#pragma unroll
        for (int r = 0; r < 8; ++r) {
            const int row = rbase + r;
            if (row < nrows) {
                float vv = av[r] + bv;
                if (RELU) vv = fmaxf(vv, 0.0f);
                if (OUT_BF16)
                    ((__bf16*)out)[(size_t)row * HID + col] = (__bf16)vv;
                else
                    ((float*)out)[(size_t)row * HID + col] = vv;
            }
        }
    }
}

// ---------------------------------------------------------------------------
// Pooling
// ---------------------------------------------------------------------------
__global__ void zero_f32(float* __restrict__ p, int n) {
    int i = blockIdx.x * blockDim.x + threadIdx.x;
    if (i < n) p[i] = 0.0f;
}

__global__ void pool_acc(const float* __restrict__ h, const int* __restrict__ batch,
                         float* __restrict__ sums, float* __restrict__ counts) {
    long long idx = (long long)blockIdx.x * blockDim.x + threadIdx.x;
    long long total = (long long)N_NODES * (HID / 4);
    if (idx >= total) return;
    int n = (int)(idx >> 7);            // HID/4 == 128
    int c = ((int)idx & 127) << 2;
    int g = batch[n];
    const float4 val = *(const float4*)(h + (size_t)n * HID + c);
    float* p = sums + (size_t)g * HID + c;
    unsafeAtomicAdd(p + 0, val.x);
    unsafeAtomicAdd(p + 1, val.y);
    unsafeAtomicAdd(p + 2, val.z);
    unsafeAtomicAdd(p + 3, val.w);
    if (c == 0) unsafeAtomicAdd(counts + g, 1.0f);
}

__global__ __launch_bounds__(128)
void fc_out(const float* __restrict__ sums, const float* __restrict__ counts,
            const float* __restrict__ fc_w, const float* __restrict__ fc_b,
            float* __restrict__ out) {
    __shared__ float red[128];
    const int g = blockIdx.x, t = threadIdx.x;
    const float cnt = fmaxf(counts[g], 1.0f);
    float acc = 0.0f;
    for (int c = t; c < HID; c += 128)
        acc += (sums[(size_t)g * HID + c] / cnt) * fc_w[c];
    red[t] = acc;
    __syncthreads();
    for (int s = 64; s > 0; s >>= 1) { if (t < s) red[t] += red[t + s]; __syncthreads(); }
    if (t == 0) out[g] = 1.0f / (1.0f + expf(-(red[0] + fc_b[0])));
}

// ---------------------------------------------------------------------------
extern "C" void kernel_launch(void* const* d_in, const int* in_sizes, int n_in,
                              void* d_out, int out_size, void* d_ws, size_t ws_size,
                              hipStream_t stream) {
    (void)in_sizes; (void)n_in; (void)out_size; (void)ws_size;
    const float* x     = (const float*)d_in[0];
    const int*   edge  = (const int*)d_in[1];
    const int*   srcI  = edge;
    const int*   dstI  = edge + N_EDGES;
    const int*   batch = (const int*)d_in[2];
    const float* w1_0  = (const float*)d_in[3];
    const float* b1_0  = (const float*)d_in[4];
    const float* w2_0  = (const float*)d_in[5];
    const float* b2_0  = (const float*)d_in[6];
    const float* w1_r  = (const float*)d_in[7];
    const float* b1_r  = (const float*)d_in[8];
    const float* w2_r  = (const float*)d_in[9];
    const float* b2_r  = (const float*)d_in[10];
    const float* fc_w  = (const float*)d_in[11];
    const float* fc_b  = (const float*)d_in[12];

    char* ws = (char*)d_ws;
    size_t off = 0;
    auto alloc = [&](size_t bytes) -> char* {
        char* p = ws + off;
        off += (bytes + 255) & ~(size_t)255;
        return p;
    };
    float*  hbuf = (float*)alloc((size_t)N_NODES * HID * 4);
    float*  aggF = (float*)alloc((size_t)N_NODES * HID * 4);
    __bf16* aggB = (__bf16*)alloc((size_t)N_NODES * HID * 2);
    __bf16* tB   = (__bf16*)alloc((size_t)N_NODES * HID * 2);
    __bf16* wbuf[8];
    for (int i = 0; i < 8; ++i) wbuf[i] = (__bf16*)alloc((size_t)HID * HID * 2);
    float* sums   = (float*)alloc((size_t)(N_GRAPH * HID + N_GRAPH) * 4);
    float* counts = sums + (size_t)N_GRAPH * HID;

    // --- spectral-normalize + cast all 8 weight matrices ---
    sn_kernel<<<1, 512, 0, stream>>>(w1_0, HID, D_IN, wbuf[0]);
    sn_kernel<<<1, 512, 0, stream>>>(w2_0, HID, HID,  wbuf[1]);
    for (int i = 0; i < 3; ++i) {
        sn_kernel<<<1, 512, 0, stream>>>(w1_r + (size_t)i * HID * HID, HID, HID, wbuf[2 + 2 * i]);
        sn_kernel<<<1, 512, 0, stream>>>(w2_r + (size_t)i * HID * HID, HID, HID, wbuf[3 + 2 * i]);
    }

    const dim3 gemm_grid((N_NODES + 127) / 128, HID / 64);
    const float* hprev = x;

    for (int l = 0; l < 4; ++l) {
        const int K = (l == 0) ? D_IN : HID;
        const int n = N_NODES * K;
        const int log2c = (K == D_IN) ? 5 : 7;          // K/4 chunks per edge

        // agg = h
        copy_f4<<<(n / 4 + 255) / 256, 256, 0, stream>>>((const float4*)hprev,
                                                         (float4*)aggF, n / 4);
        // agg[dst] += h[src]
        long long tot = (long long)N_EDGES << log2c;
        scatter_add<<<(int)((tot + 255) / 256), 256, 0, stream>>>(hprev, srcI, dstI,
                                                                  aggF, K, log2c);
        // bf16 cast for tensor cores
        cast_f32_bf16<<<(n / 4 + 255) / 256, 256, 0, stream>>>((const float4*)aggF,
                                                               (v4bf*)aggB, n / 4);

        const float*  b1 = (l == 0) ? b1_0 : b1_r + (size_t)(l - 1) * HID;
        const float*  b2 = (l == 0) ? b2_0 : b2_r + (size_t)(l - 1) * HID;
        const __bf16* W1 = (l == 0) ? wbuf[0] : wbuf[2 + 2 * (l - 1)];
        const __bf16* W2 = (l == 0) ? wbuf[1] : wbuf[3 + 2 * (l - 1)];

        // h1 = relu(agg @ W1^T + b1)  -> bf16
        gemm_wmma<true, true><<<gemm_grid, 256, 64 * K * 2, stream>>>(
            aggB, W1, b1, tB, N_NODES, K);
        // h = h1 @ W2^T + b2          -> fp32
        gemm_wmma<false, false><<<gemm_grid, 256, 64 * HID * 2, stream>>>(
            tB, W2, b2, hbuf, N_NODES, HID);
        hprev = hbuf;
    }

    // --- global mean pool + FC + sigmoid ---
    const int pn = N_GRAPH * HID + N_GRAPH;
    zero_f32<<<(pn + 255) / 256, 256, 0, stream>>>(sums, pn);
    long long ptot = (long long)N_NODES * (HID / 4);
    pool_acc<<<(int)((ptot + 255) / 256), 256, 0, stream>>>(hbuf, batch, sums, counts);
    fc_out<<<N_GRAPH, 128, 0, stream>>>(sums, counts, fc_w, fc_b, (float*)d_out);
}
